// MovingAverageGatedAttention_77678778515531
// MI455X (gfx1250) — compile-verified, matrix operand order checked by reading, not run on hardware
//
#include <hip/hip_runtime.h>
#include <hip/hip_bf16.h>
#include <math.h>

// ---- problem constants ----
#define L_SEQ   2048
#define B_SZ    8
#define D_MODEL 1024
#define Z_DIM   128
#define H_DIM   2048
#define N_EMA   16
#define NOUT    (Z_DIM + H_DIM + 2 * D_MODEL)   // 4224
#define NLB     (L_SEQ * B_SZ)                  // 16384
#define QK      (2 * Z_DIM)                     // 256 : [q|rot(a)] , [k|rot(b)]

typedef __attribute__((ext_vector_type(16))) __bf16 v16bf;
typedef __attribute__((ext_vector_type(8)))  float  v8f;

__device__ __forceinline__ float sigm_(float x) { return 1.0f / (1.0f + __expf(-x)); }
__device__ __forceinline__ float silu_(float x) { return x * sigm_(x); }
__device__ __forceinline__ __hip_bfloat16 f2bf(float x) { return __float2bfloat16(x); }
__device__ __forceinline__ float bf2f(__hip_bfloat16 x) { return __bfloat162float(x); }

// Async DMA: 32 bytes per lane (two b128 ops, INST_OFFSET applies to both the
// LDS and global side), tracked on ASYNCcnt, no VGPR staging.
__device__ __forceinline__ void async_copy_32B(unsigned lds_off, const void* gsrc) {
    asm volatile("global_load_async_to_lds_b128 %0, %1, off\n\t"
                 "global_load_async_to_lds_b128 %0, %1, off offset:16"
                 :: "v"(lds_off), "v"(gsrc) : "memory");
}

// =====================================================================
// f32 -> bf16 weight conversion (grid-stride)
// =====================================================================
__global__ void k_f32_to_bf16(const float* __restrict__ in,
                              __hip_bfloat16* __restrict__ out, int n) {
    int i = blockIdx.x * blockDim.x + threadIdx.x;
    int stride = gridDim.x * blockDim.x;
    for (; i < n; i += stride) out[i] = f2bf(in[i]);
}

// =====================================================================
// LayerNorm over D=1024. One block (256 thr) per (l,b) row. Output bf16.
// =====================================================================
__global__ void k_layernorm(const float* __restrict__ x,
                            const float* __restrict__ w,
                            const float* __restrict__ b,
                            __hip_bfloat16* __restrict__ xn) {
    __shared__ float s1[256];
    __shared__ float s2[256];
    const int row = blockIdx.x;
    const float* xr = x + (size_t)row * D_MODEL;
    float v[4]; float sum = 0.f, sq = 0.f;
#pragma unroll
    for (int j = 0; j < 4; j++) {
        v[j] = xr[j * 256 + threadIdx.x];
        sum += v[j]; sq += v[j] * v[j];
    }
    s1[threadIdx.x] = sum; s2[threadIdx.x] = sq;
    __syncthreads();
    for (int o = 128; o > 0; o >>= 1) {
        if (threadIdx.x < o) {
            s1[threadIdx.x] += s1[threadIdx.x + o];
            s2[threadIdx.x] += s2[threadIdx.x + o];
        }
        __syncthreads();
    }
    const float mu  = s1[0] * (1.0f / D_MODEL);
    const float var = s2[0] * (1.0f / D_MODEL) - mu * mu;
    const float inv = rsqrtf(var + 1e-5f);
#pragma unroll
    for (int j = 0; j < 4; j++) {
        int c = j * 256 + threadIdx.x;
        xn[(size_t)row * D_MODEL + c] = f2bf((v[j] - mu) * inv * w[c] + b[c]);
    }
}

// =====================================================================
// EMA as N=16 parallel 1st-order IIR scans per channel.
// kern[l] = sum_n p_n*beta_n*q_n^l  ->  s_n[t] = q_n*s_n[t-1] + x[t]
// One thread per (b,d). Output mx = silu(conv + xn*omega) as bf16.
// =====================================================================
__global__ void k_ema(const __hip_bfloat16* __restrict__ xn,
                      const float* __restrict__ delta,
                      const float* __restrict__ alpha,
                      const float* __restrict__ beta,
                      const float* __restrict__ gamma,
                      const float* __restrict__ omega,
                      __hip_bfloat16* __restrict__ mx) {
    const int g = blockIdx.x * blockDim.x + threadIdx.x; // B*D = 8192
    const int d = g % D_MODEL;
    const int b = g / D_MODEL;
    float q[N_EMA], c[N_EMA], s[N_EMA];
#pragma unroll
    for (int n = 0; n < N_EMA; n++) {
        const int pi = d * N_EMA + n;
        const float p = sigm_(delta[pi]);
        const float a = sigm_(alpha[pi]);
        q[n] = 1.0f - p * a;
        c[n] = p * beta[pi] * gamma[pi] * 0.25f;   // scale = sqrt(1/16)
        s[n] = 0.0f;
    }
    const float om = omega[d];
    for (int t = 0; t < L_SEQ; t++) {
        const size_t idx = ((size_t)t * B_SZ + b) * D_MODEL + d;
        const float xv = bf2f(xn[idx]);
        float acc = 0.0f;
#pragma unroll
        for (int n = 0; n < N_EMA; n++) {
            s[n] = fmaf(q[n], s[n], xv);
            acc  = fmaf(c[n], s[n], acc);
        }
        mx[idx] = f2bf(silu_(acc + xv * om));
    }
}

// =====================================================================
// Rotary bias rows: bias = rot(alpha) @ rot(beta)^T (rank-Z) folded into
// the QK gemm by appending rot(alpha) to q' and rot(beta) to k'.
// =====================================================================
__global__ void k_rot(const float* __restrict__ ra_w,
                      const float* __restrict__ rb_w,
                      __hip_bfloat16* __restrict__ qp,
                      __hip_bfloat16* __restrict__ kp) {
    const int g = blockIdx.x * blockDim.x + threadIdx.x; // L*Z
    const int z = g % Z_DIM;
    const int l = g / Z_DIM;
    const int zi = z & 63;
    const float freq = __expf(-0.14391156831f * (float)zi); // ln(10000)/64
    float sn, cs;
    __sincosf((float)l * freq, &sn, &cs);
    float ra, rb;
    if (z < 64) {
        ra = ra_w[z] * cs - ra_w[z + 64] * sn;
        rb = rb_w[z] * cs - rb_w[z + 64] * sn;
    } else {
        ra = ra_w[z] * cs + ra_w[z - 64] * sn;
        rb = rb_w[z] * cs + rb_w[z - 64] * sn;
    }
#pragma unroll
    for (int b = 0; b < B_SZ; b++) {
        const size_t base = ((size_t)b * L_SEQ + l) * QK + Z_DIM + z;
        qp[base] = f2bf(ra);
        kp[base] = f2bf(rb);
    }
}

// =====================================================================
// Generic bf16 WMMA GEMM:  C[m,n] = sum_k A[m,k] * B'[n,k]
//   BT=1 : B given as (N,K) row-major  -> async DMA straight into LDS
//   BT=0 : B given as (K,N) row-major  -> register transpose into LDS
// Block 256 thr = 8 waves (4m x 2n), tile 128x128x32, double-buffered LDS.
// A tiles (and NT B tiles) use GLOBAL_LOAD_ASYNC_TO_LDS_B128 (ASYNCcnt),
// overlapping the DMA of tile t+1 with the WMMA stream of tile t.
// =====================================================================
template <bool BT, typename Epi>
__global__ __launch_bounds__(256) void k_gemm(const __hip_bfloat16* __restrict__ A,
                                              const __hip_bfloat16* __restrict__ Bm,
                                              int M, int N, int K,
                                              size_t strideA, size_t strideB,
                                              Epi epi) {
    constexpr int BM = 128, BN = 128, BK = 32;
    constexpr int PAD = 48; // halves per LDS row: 96B (32B-aligned, conflict-reducing)
    constexpr int NB_ASYNC = BT ? 4 : 2; // async ops issued per wave per tile
    __shared__ alignas(32) unsigned short As[2][BM * PAD];
    __shared__ alignas(32) unsigned short Bs[2][BN * PAD];

    const int tid  = threadIdx.x;
    const int lane = tid & 31;
    const int wave = tid >> 5;
    const int wm = (wave & 3) * 32;  // wave m-offset
    const int wn = (wave >> 2) * 64; // wave n-offset
    const int m0 = blockIdx.y * BM;
    const int n0 = blockIdx.x * BN;
    const int zb = blockIdx.z;

    const unsigned short* Ag = (const unsigned short*)(A  + (size_t)zb * strideA);
    const unsigned short* Bg = (const unsigned short*)(Bm + (size_t)zb * strideB);

    v8f acc[2][4];
#pragma unroll
    for (int i = 0; i < 2; i++)
#pragma unroll
        for (int j = 0; j < 4; j++) acc[i][j] = (v8f){};

    const int arow  = tid >> 1;          // 0..127
    const int ahalf = (tid & 1) * 16;    // 0 or 16 halves
    const int bkrow = tid >> 3;          // 0..31  (BT=0 path)
    const int bncol = (tid & 7) * 16;    // 0..112 (BT=0 path)

    // stage tile starting at column kt into LDS buffer `buf`
    auto stage = [&](int kt, int buf) {
        async_copy_32B((unsigned)(size_t)&As[buf][arow * PAD + ahalf],
                       Ag + (size_t)(m0 + arow) * K + kt + ahalf);
        if (BT) {
            async_copy_32B((unsigned)(size_t)&Bs[buf][arow * PAD + ahalf],
                           Bg + (size_t)(n0 + arow) * K + kt + ahalf);
        } else {
            union { uint4 u[2]; unsigned short s[16]; } t;
            const unsigned short* src = Bg + (size_t)(kt + bkrow) * N + n0 + bncol;
            t.u[0] = *(const uint4*)(src);
            t.u[1] = *(const uint4*)(src + 8);
#pragma unroll
            for (int i = 0; i < 16; i++) Bs[buf][(bncol + i) * PAD + bkrow] = t.s[i];
            if (kt + BK < K)  // pull the K-tile after next toward the caches
                __builtin_prefetch(src + (size_t)BK * N, 0, 3);
        }
    };

    const int ntiles = K / BK;
    stage(0, 0);

    for (int t = 0; t < ntiles; t++) {
        const int buf = t & 1;
        if (t + 1 < ntiles) {
            stage((t + 1) * BK, buf ^ 1);
            // async ops complete in order: once only tile t+1's NB_ASYNC remain
            // outstanding, tile t is fully resident in LDS.
            asm volatile("s_wait_asynccnt %0" :: "n"(NB_ASYNC) : "memory");
        } else {
            asm volatile("s_wait_asynccnt 0" ::: "memory");
        }
        __syncthreads();

        // ---- WMMA: lanes 0-15 hold K[0:16), lanes 16-31 K[16:32) ----
        const int fr = lane & 15;
        const int kh = (lane >> 4) * 16;
        v16bf afr[2], bfr[4];
#pragma unroll
        for (int i = 0; i < 2; i++)
            afr[i] = *(const v16bf*)(&As[buf][(wm + i * 16 + fr) * PAD + kh]);
#pragma unroll
        for (int j = 0; j < 4; j++)
            bfr[j] = *(const v16bf*)(&Bs[buf][(wn + j * 16 + fr) * PAD + kh]);
#pragma unroll
        for (int i = 0; i < 2; i++)
#pragma unroll
            for (int j = 0; j < 4; j++)
                acc[i][j] = __builtin_amdgcn_wmma_f32_16x16x32_bf16(
                    false, afr[i], false, bfr[j], (short)0, acc[i][j], false, false);
        __syncthreads(); // buffer buf is re-staged next iteration
    }

    // ---- epilogue: C 16x16 f32 layout: lanes<16 -> M=vgpr, lanes>=16 -> M=vgpr+8 ----
    const int fr = lane & 15;
    const int mo = (lane >> 4) * 8;
#pragma unroll
    for (int i = 0; i < 2; i++)
#pragma unroll
        for (int j = 0; j < 4; j++)
#pragma unroll
            for (int e = 0; e < 8; e++) {
                const int m = m0 + wm + i * 16 + mo + e;
                const int n = n0 + wn + j * 16 + fr;
                epi(m, n, zb, acc[i][j][e]);
            }
}

// ---- epilogues ----
struct EpiV {  // v = silu(xn@v_w^T + v_b) stored as (B,L,H) bf16
    const float* vb; __hip_bfloat16* v_blh;
    __device__ void operator()(int m, int n, int, float v) const {
        const int b = m % B_SZ, l = m / B_SZ;
        v_blh[((size_t)b * L_SEQ + l) * H_DIM + n] = f2bf(silu_(v + vb[n]));
    }
};

struct EpiBase {  // base = mx@mx_w^T + mx_b, split into u | z->q,k | r | hx
    const float* mxb; const float* qkg; const float* qkb;
    float* u; __hip_bfloat16* qp; __hip_bfloat16* kp;
    __hip_bfloat16* r; float* hx;
    __device__ void operator()(int m, int n, int, float v) const {
        v += mxb[n];
        if (n < D_MODEL) {
            u[(size_t)m * D_MODEL + n] = sigm_(v);
        } else if (n < D_MODEL + Z_DIM) {
            const int zz = n - D_MODEL;
            const float zs = silu_(v);
            const int b = m % B_SZ, l = m / B_SZ;
            const float qv = (zs * qkg[zz] + qkb[zz]) * 0.08838834765f; // Z^-0.5
            const float kv =  zs * qkg[Z_DIM + zz] + qkb[Z_DIM + zz];
            const size_t base = ((size_t)b * L_SEQ + l) * QK + zz;
            qp[base] = f2bf(qv);
            kp[base] = f2bf(kv);
        } else if (n < D_MODEL + Z_DIM + H_DIM) {
            r[(size_t)m * H_DIM + (n - D_MODEL - Z_DIM)] = f2bf(silu_(v));
        } else {
            hx[(size_t)m * D_MODEL + (n - D_MODEL - Z_DIM - H_DIM)] = v;
        }
    }
};

struct EpiScores {  // raw attention logits (bias already folded via rank-Z concat)
    float* sc;
    __device__ void operator()(int m, int n, int zb, float v) const {
        sc[((size_t)zb * L_SEQ + m) * L_SEQ + n] = v;
    }
};

struct EpiAttnV {  // h[l,b,h] * r[l,b,h] -> bf16 for the final GEMM
    const __hip_bfloat16* r; __hip_bfloat16* hr;
    __device__ void operator()(int m, int n, int zb, float v) const {
        const size_t idx = ((size_t)m * B_SZ + zb) * H_DIM + n;
        hr[idx] = f2bf(v * bf2f(r[idx]));
    }
};

struct EpiOut {  // h=silu(hx + (h*r)@h_w^T + h_b); out = res + u*(h-res)
    const float* hb; const float* hx; const float* u; const float* x; float* out;
    __device__ void operator()(int m, int n, int, float v) const {
        const size_t idx = (size_t)m * D_MODEL + n;
        const float h   = silu_(hx[idx] + v + hb[n]);
        const float res = x[idx];
        out[idx] = res + u[idx] * (h - res);
    }
};

// =====================================================================
// Row softmax over L=2048 (f32 in, bf16 probs out). One block per row.
// =====================================================================
__global__ void k_softmax(const float* __restrict__ sc,
                          __hip_bfloat16* __restrict__ pr) {
    __shared__ float red[256];
    const size_t row = blockIdx.x; // B*L rows
    const float* s = sc + row * L_SEQ;
    __hip_bfloat16* p = pr + row * L_SEQ;
    float v[8]; float mx = -1e30f;
#pragma unroll
    for (int j = 0; j < 8; j++) {
        v[j] = s[j * 256 + threadIdx.x];
        mx = fmaxf(mx, v[j]);
    }
    red[threadIdx.x] = mx; __syncthreads();
    for (int o = 128; o > 0; o >>= 1) {
        if (threadIdx.x < o) red[threadIdx.x] = fmaxf(red[threadIdx.x], red[threadIdx.x + o]);
        __syncthreads();
    }
    mx = red[0]; __syncthreads();
    float sum = 0.f;
#pragma unroll
    for (int j = 0; j < 8; j++) { v[j] = __expf(v[j] - mx); sum += v[j]; }
    red[threadIdx.x] = sum; __syncthreads();
    for (int o = 128; o > 0; o >>= 1) {
        if (threadIdx.x < o) red[threadIdx.x] += red[threadIdx.x + o];
        __syncthreads();
    }
    const float inv = 1.0f / red[0];
#pragma unroll
    for (int j = 0; j < 8; j++) p[j * 256 + threadIdx.x] = f2bf(v[j] * inv);
}

// =====================================================================
extern "C" void kernel_launch(void* const* d_in, const int* in_sizes, int n_in,
                              void* d_out, int out_size, void* d_ws, size_t ws_size,
                              hipStream_t stream) {
    const float* x        = (const float*)d_in[0];
    const float* ln_w     = (const float*)d_in[1];
    const float* ln_b     = (const float*)d_in[2];
    const float* v_w      = (const float*)d_in[3];
    const float* v_b      = (const float*)d_in[4];
    const float* mx_w     = (const float*)d_in[5];
    const float* mx_b     = (const float*)d_in[6];
    const float* h_w      = (const float*)d_in[7];
    const float* h_b      = (const float*)d_in[8];
    const float* qk_gamma = (const float*)d_in[9];
    const float* qk_beta  = (const float*)d_in[10];
    const float* e_delta  = (const float*)d_in[11];
    const float* e_alpha  = (const float*)d_in[12];
    const float* e_beta   = (const float*)d_in[13];
    const float* e_gamma  = (const float*)d_in[14];
    const float* e_omega  = (const float*)d_in[15];
    const float* rb_alpha = (const float*)d_in[16];
    const float* rb_beta  = (const float*)d_in[17];
    float* out = (float*)d_out;

    // ---- workspace carve-out ----
    char* w = (char*)d_ws;
    size_t off = 0;
    auto carve = [&](size_t bytes) -> char* {
        char* p = w + off;
        off += (bytes + 255) & ~(size_t)255;
        return p;
    };
    __hip_bfloat16* xn_bf  = (__hip_bfloat16*)carve((size_t)NLB * D_MODEL * 2);
    __hip_bfloat16* mx_bf  = (__hip_bfloat16*)carve((size_t)NLB * D_MODEL * 2);
    __hip_bfloat16* vw_bf  = (__hip_bfloat16*)carve((size_t)H_DIM * D_MODEL * 2);
    __hip_bfloat16* mxw_bf = (__hip_bfloat16*)carve((size_t)NOUT * D_MODEL * 2);
    __hip_bfloat16* hw_bf  = (__hip_bfloat16*)carve((size_t)D_MODEL * H_DIM * 2);
    __hip_bfloat16* v_blh  = (__hip_bfloat16*)carve((size_t)B_SZ * L_SEQ * H_DIM * 2);
    float*          u_buf  = (float*)carve((size_t)NLB * D_MODEL * 4);
    __hip_bfloat16* qp     = (__hip_bfloat16*)carve((size_t)B_SZ * L_SEQ * QK * 2);
    __hip_bfloat16* kp     = (__hip_bfloat16*)carve((size_t)B_SZ * L_SEQ * QK * 2);
    __hip_bfloat16* r_buf  = (__hip_bfloat16*)carve((size_t)NLB * H_DIM * 2);
    float*          hx_buf = (float*)carve((size_t)NLB * D_MODEL * 4);
    float*          scores = (float*)carve((size_t)B_SZ * L_SEQ * L_SEQ * 4);
    __hip_bfloat16* probs  = (__hip_bfloat16*)carve((size_t)B_SZ * L_SEQ * L_SEQ * 2);
    __hip_bfloat16* hr     = (__hip_bfloat16*)carve((size_t)NLB * H_DIM * 2);
    (void)ws_size; (void)n_in; (void)in_sizes; (void)out_size;

    // ---- 0) weights -> bf16 ----
    k_f32_to_bf16<<<1024, 256, 0, stream>>>(v_w,  vw_bf,  H_DIM * D_MODEL);
    k_f32_to_bf16<<<1024, 256, 0, stream>>>(mx_w, mxw_bf, NOUT * D_MODEL);
    k_f32_to_bf16<<<1024, 256, 0, stream>>>(h_w,  hw_bf,  D_MODEL * H_DIM);

    // ---- 1) LayerNorm ----
    k_layernorm<<<NLB, 256, 0, stream>>>(x, ln_w, ln_b, xn_bf);

    // ---- 2) EMA scan ----
    k_ema<<<(B_SZ * D_MODEL) / 256, 256, 0, stream>>>(xn_bf, e_delta, e_alpha,
                                                      e_beta, e_gamma, e_omega, mx_bf);

    // ---- 3) rotary rows into q'/k' tails ----
    k_rot<<<(L_SEQ * Z_DIM) / 256, 256, 0, stream>>>(rb_alpha, rb_beta, qp, kp);

    // ---- 4) v = silu(xn @ v_w^T + v_b) ----
    k_gemm<true, EpiV><<<dim3(H_DIM / 128, NLB / 128, 1), 256, 0, stream>>>(
        xn_bf, vw_bf, NLB, H_DIM, D_MODEL, 0, 0, EpiV{v_b, v_blh});

    // ---- 5) base = mx @ mx_w^T + mx_b, fused split epilogue ----
    k_gemm<true, EpiBase><<<dim3(NOUT / 128, NLB / 128, 1), 256, 0, stream>>>(
        mx_bf, mxw_bf, NLB, NOUT, D_MODEL, 0, 0,
        EpiBase{mx_b, qk_gamma, qk_beta, u_buf, qp, kp, r_buf, hx_buf});

    // ---- 6) scores = q' @ k'^T (bias folded), batched over B ----
    k_gemm<true, EpiScores><<<dim3(L_SEQ / 128, L_SEQ / 128, B_SZ), 256, 0, stream>>>(
        qp, kp, L_SEQ, L_SEQ, QK, (size_t)L_SEQ * QK, (size_t)L_SEQ * QK,
        EpiScores{scores});

    // ---- 7) softmax ----
    k_softmax<<<B_SZ * L_SEQ, 256, 0, stream>>>(scores, probs);

    // ---- 8) h = probs @ v  (NN: LDS transpose path), fused *r ----
    k_gemm<false, EpiAttnV><<<dim3(H_DIM / 128, L_SEQ / 128, B_SZ), 256, 0, stream>>>(
        probs, v_blh, L_SEQ, H_DIM, L_SEQ, (size_t)L_SEQ * L_SEQ,
        (size_t)L_SEQ * H_DIM, EpiAttnV{r_buf, hr});

    // ---- 9) out = res + u * (silu(hx + hr@h_w^T + h_b) - res) ----
    k_gemm<true, EpiOut><<<dim3(D_MODEL / 128, NLB / 128, 1), 256, 0, stream>>>(
        hr, hw_bf, NLB, D_MODEL, H_DIM, 0, 0,
        EpiOut{h_b, hx_buf, u_buf, x, out});
}
